// GNNLayer_74577812128000
// MI455X (gfx1250) — matrix operand order repaired and checked by reading, not compiled, not run on hardware
//
#include <hip/hip_runtime.h>

#define NN 10000
#define NE 300000
#define DIM 256
#define WPITCH (DIM + 8)   // LDS row pitch in halves: 528 bytes = 132 dwords (bank-friendly)

typedef __attribute__((ext_vector_type(16))) _Float16 v16h;
typedef __attribute__((ext_vector_type(8)))  _Float16 v8h;
typedef __attribute__((ext_vector_type(8)))  float    v8f;
typedef __attribute__((ext_vector_type(4)))  float    v4f;
typedef __attribute__((ext_vector_type(4)))  unsigned int u32x4;
typedef __attribute__((ext_vector_type(8)))  int      i32x8;
typedef __attribute__((ext_vector_type(4)))  int      i32x4;

#ifndef __has_builtin
#define __has_builtin(x) 0
#endif
#if __has_builtin(__builtin_amdgcn_tensor_load_to_lds) && __has_builtin(__builtin_amdgcn_s_wait_tensorcnt)
#define USE_TDM 1
#else
#define USE_TDM 0
#endif
#if __has_builtin(__builtin_amdgcn_global_load_async_to_lds_b128)
#define USE_ASYNC 1
#else
#define USE_ASYNC 0
#endif

// ---------------------------------------------------------------------------
// Stage the 256x256 f16 weight matrix (128KB) into LDS with a 528B row pitch.
// One Tensor Data Mover descriptor (2D tile, pad 4 dwords per 128 dwords ->
// exactly 16B padding after each 512B row), issued by wave 0.
// ---------------------------------------------------------------------------
__device__ __forceinline__ void stage_weights(const _Float16* __restrict__ W,
                                              _Float16* sW) {
#if USE_TDM
  if (threadIdx.x < 32) {
    const unsigned lds_base = (unsigned)(size_t)(void*)sW;
    const unsigned long long ga = (unsigned long long)(size_t)(const void*)W;
    u32x4 g0;
    g0[0] = 1u;                                   // count=1, user descriptor
    g0[1] = lds_base;                             // lds_addr
    g0[2] = (unsigned)(ga & 0xFFFFFFFFu);         // global_addr[31:0]
    g0[3] = (unsigned)((ga >> 32) & 0x01FFFFFFu)  // global_addr[56:32]
          | (2u << 30);                           // type = 2 ("image")
    i32x8 g1;
    g1[0] = (1 << 16)      // data_size = 1 -> 2 bytes
          | (1 << 20)      // pad_enable
          | (6 << 22)      // pad_interval code 6 -> every 128 dwords (512B row)
          | (3 << 25);     // pad_amount code 3 -> 4 dwords (16B)
    g1[1] = (DIM & 0xFFFF) << 16;   // tensor_dim0[15:0] = 256
    g1[2] = (DIM & 0xFFFF) << 16;   // tensor_dim0[31:16]=0 | tensor_dim1[15:0]=256
    g1[3] = (DIM & 0xFFFF) << 16;   // tensor_dim1[31:16]=0 | tile_dim0 = 256
    g1[4] = DIM;                    // tile_dim1 = 256, tile_dim2 = 0
    g1[5] = DIM;                    // tensor_dim0_stride[31:0] = 256
    g1[6] = 0;                      // stride0[47:32]=0 | stride1[15:0]=0
    g1[7] = 0;
    i32x4 g2; g2[0] = 0; g2[1] = 0; g2[2] = 0; g2[3] = 0;   // 2D: unused
    i32x4 g3; g3[0] = 0; g3[1] = 0; g3[2] = 0; g3[3] = 0;
    i32x8 g4;                                                // extra group (zero)
    g4[0] = 0; g4[1] = 0; g4[2] = 0; g4[3] = 0;
    g4[4] = 0; g4[5] = 0; g4[6] = 0; g4[7] = 0;
    __builtin_amdgcn_tensor_load_to_lds(g0, g1, g2, g3, g4, 0);
    __builtin_amdgcn_s_wait_tensorcnt(0);
  }
#elif USE_ASYNC
  for (int c = threadIdx.x; c < (DIM * DIM) / 8; c += 256) {
    const int r = c >> 5, cc = c & 31;             // row, 16B-chunk within row
    const _Float16* g = W + (size_t)c * 8;
    _Float16* l = (_Float16*)((char*)sW + (size_t)r * (WPITCH * 2) + cc * 16);
    __builtin_amdgcn_global_load_async_to_lds_b128(
        (const __attribute__((address_space(1))) void*)g,
        (__attribute__((address_space(3))) void*)l, 0, 0);
  }
#if __has_builtin(__builtin_amdgcn_s_wait_asynccnt)
  __builtin_amdgcn_s_wait_asynccnt(0);
#else
  asm volatile("s_wait_asynccnt 0x0" ::: "memory");
#endif
#else
  for (int c = threadIdx.x; c < (DIM * DIM) / 8; c += 256) {
    const int r = c >> 5, cc = c & 31;
    *(v8h*)((char*)sW + (size_t)r * (WPITCH * 2) + cc * 16) =
        *(const v8h*)(W + (size_t)c * 8);
  }
#endif
  __syncthreads();
}

// ---------------------------------------------------------------------------
// Weight fp32 -> f16 conversion (5 matrices of 256x256, kept [j,k] row-major).
// ---------------------------------------------------------------------------
__global__ void k_cvt_w(const float* __restrict__ u, const float* __restrict__ v,
                        const float* __restrict__ a, const float* __restrict__ b,
                        const float* __restrict__ c, _Float16* __restrict__ out) {
  int i = blockIdx.x * blockDim.x + threadIdx.x;   // [0, 5*65536)
  int m = i >> 16, j = i & 0xFFFF;
  const float* s = (m == 0) ? u : (m == 1) ? v : (m == 2) ? a : (m == 3) ? b : c;
  out[i] = (_Float16)s[j];
}

__global__ void k_zero(float* __restrict__ p, int n) {
  int i = blockIdx.x * blockDim.x + threadIdx.x;
  if (i < n) p[i] = 0.f;
}

// ---------------------------------------------------------------------------
// One wave computes a 16(M) x 256(N) strip, K=256, software-pipelined:
//  - A (fp32 activations, global) double-buffered across k-steps
//  - B (f16 weights, padded LDS)  double-buffered across n-tiles
// A layout (16-bit A 16x32): lanes<16 -> M=L, halves K{0..7,16..23};
// lanes>=16 -> M=L-16, halves K{8..15,24..31}.
// ---------------------------------------------------------------------------
__device__ __forceinline__ void load_a_raw(const float* __restrict__ xr, int k0,
                                           int halfsel, v4f ar[4]) {
  const float* ap = xr + k0 + 8 * halfsel;
  ar[0] = *(const v4f*)(ap + 0);
  ar[1] = *(const v4f*)(ap + 4);
  ar[2] = *(const v4f*)(ap + 16);
  ar[3] = *(const v4f*)(ap + 20);
}

__device__ __forceinline__ void wmma_rowstrip(const float* __restrict__ X,
                                              const _Float16* sW,
                                              int r0, int lane, int nrows,
                                              v8f acc[16]) {
  const int lane16 = lane & 15;
  const int halfsel = lane >> 4;
  int row = r0 + lane16;
  row = (row < nrows) ? row : (nrows - 1);   // clamp; OOB handled at store
  const float* __restrict__ xr = X + (size_t)row * DIM;

  v4f arc[4], arn[4];
  load_a_raw(xr, 0, halfsel, arc);           // prologue: A for k0 = 0

  for (int k0 = 0; k0 < DIM; k0 += 32) {
    if (k0 + 32 < DIM) {
      load_a_raw(xr, k0 + 32, halfsel, arn); // next A in flight during this k-step
      __builtin_prefetch(xr + k0 + 64, 0, 0);
    }
    v16h A;
#pragma unroll
    for (int i = 0; i < 4; ++i) {
      A[i]      = (_Float16)arc[0][i];
      A[4 + i]  = (_Float16)arc[1][i];
      A[8 + i]  = (_Float16)arc[2][i];
      A[12 + i] = (_Float16)arc[3][i];
    }
    // B double buffer over the 16 n-tiles
    const _Float16* bb = sW + (size_t)lane16 * WPITCH + k0 + 8 * halfsel;
    v8h b0 = *(const v8h*)(bb);
    v8h b1 = *(const v8h*)(bb + 16);
#pragma unroll
    for (int t = 0; t < 16; ++t) {
      v8h n0 = b0, n1 = b1;
      if (t < 15) {
        const _Float16* nb = bb + (size_t)(t + 1) * 16 * WPITCH;
        n0 = *(const v8h*)(nb);
        n1 = *(const v8h*)(nb + 16);
      }
      v16h B;
#pragma unroll
      for (int i = 0; i < 8; ++i) { B[i] = b0[i]; B[8 + i] = b1[i]; }
      acc[t] = __builtin_amdgcn_wmma_f32_16x16x32_f16(
          false, A, false, B, (short)0, acc[t], false, false);
      b0 = n0; b1 = n1;
    }
#pragma unroll
    for (int i = 0; i < 4; ++i) arc[i] = arn[i];
  }
}

// ---------------------------------------------------------------------------
// Node GEMMs: z selects U/V/B/C.  out = h @ w^T + b
// ---------------------------------------------------------------------------
__global__ void __launch_bounds__(256) k_gemm_node(
    const float* __restrict__ H, const _Float16* __restrict__ W16,
    const float* __restrict__ bU, const float* __restrict__ bV,
    const float* __restrict__ bB, const float* __restrict__ bC,
    float* __restrict__ oU, float* __restrict__ oV,
    float* __restrict__ oB, float* __restrict__ oC) {
  __shared__ _Float16 sW[DIM * WPITCH];        // 135,168 B
  const int z  = blockIdx.z;                   // 0:U 1:V 2:B 3:C
  const int mm = (z < 2) ? z : z + 2;          // slot in packed f16 weights
  const _Float16* W = W16 + (size_t)mm * DIM * DIM;
  const float* bias = (z == 0) ? bU : (z == 1) ? bV : (z == 2) ? bB : bC;
  float* out        = (z == 0) ? oU : (z == 1) ? oV : (z == 2) ? oB : oC;

  stage_weights(W, sW);

  const int wave = threadIdx.x >> 5, lane = threadIdx.x & 31;
  const int lane16 = lane & 15, halfsel = lane >> 4;
  const int r0 = blockIdx.x * 128 + wave * 16;

  v8f acc[16];
#pragma unroll
  for (int t = 0; t < 16; ++t)
#pragma unroll
    for (int i = 0; i < 8; ++i) acc[t][i] = 0.f;

  wmma_rowstrip(H, sW, r0, lane, NN, acc);

#pragma unroll
  for (int t = 0; t < 16; ++t) {
    const int col = t * 16 + lane16;
    const float b = bias[col];
#pragma unroll
    for (int r = 0; r < 8; ++r) {
      const int row = r0 + r + 8 * halfsel;    // C/D layout: VGPR r -> M=r+8*halfsel
      if (row < NN) out[(size_t)row * DIM + col] = acc[t][r] + b;
    }
  }
}

// ---------------------------------------------------------------------------
// Edge GEMM fused with gather-add and BN column statistics:
//   t = e @ A_w^T + A_b + Bh[src] + Ch[dst] -> e_out region (scratch)
// ---------------------------------------------------------------------------
__global__ void __launch_bounds__(256) k_gemm_edge(
    const float* __restrict__ E, const _Float16* __restrict__ WA,
    const float* __restrict__ bA, const int* __restrict__ ei,
    const float* __restrict__ Bh, const float* __restrict__ Ch,
    float* __restrict__ T, float* __restrict__ colsum, float* __restrict__ colsq) {
  __shared__ _Float16 sW[DIM * WPITCH];
  __shared__ float ssum[DIM];
  __shared__ float ssq[DIM];
  if (threadIdx.x < DIM) { ssum[threadIdx.x] = 0.f; ssq[threadIdx.x] = 0.f; }

  stage_weights(WA, sW);   // includes __syncthreads()

  const int wave = threadIdx.x >> 5, lane = threadIdx.x & 31;
  const int lane16 = lane & 15, halfsel = lane >> 4;
  const int r0 = blockIdx.x * 128 + wave * 16;

  v8f acc[16];
#pragma unroll
  for (int t = 0; t < 16; ++t)
#pragma unroll
    for (int i = 0; i < 8; ++i) acc[t][i] = 0.f;

  wmma_rowstrip(E, sW, r0, lane, NE, acc);

  int  srcs[8], dsts[8];
  bool valid[8];
#pragma unroll
  for (int r = 0; r < 8; ++r) {
    const int row = r0 + r + 8 * halfsel;
    valid[r] = (row < NE);
    const int rc = valid[r] ? row : 0;
    srcs[r] = ei[rc];
    dsts[r] = ei[NE + rc];
  }

#pragma unroll
  for (int t = 0; t < 16; ++t) {
    const int col = t * 16 + lane16;
    const float b = bA[col];
    float ps = 0.f, pq = 0.f;
#pragma unroll
    for (int r = 0; r < 8; ++r) {
      if (valid[r]) {
        const int row = r0 + r + 8 * halfsel;
        const float v = acc[t][r] + b
                      + Bh[(size_t)srcs[r] * DIM + col]
                      + Ch[(size_t)dsts[r] * DIM + col];
        T[(size_t)row * DIM + col] = v;
        ps += v;
        pq += v * v;
      }
    }
    atomicAdd(&ssum[col], ps);
    atomicAdd(&ssq[col], pq);
  }
  __syncthreads();
  if (threadIdx.x < DIM) {
    atomicAdd(&colsum[threadIdx.x], ssum[threadIdx.x]);
    atomicAdd(&colsq[threadIdx.x],  ssq[threadIdx.x]);
  }
}

// ---------------------------------------------------------------------------
// Messages: msg = sigmoid(e) * Vh[dst]; scatter-sum onto agg[src].
// ---------------------------------------------------------------------------
__global__ void k_scatter(const float* __restrict__ E, const float* __restrict__ Vh,
                          const int* __restrict__ ei, float* __restrict__ agg) {
  const long long idx = (long long)blockIdx.x * blockDim.x + threadIdx.x;
  const int e4 = (int)(idx >> 6), c4 = (int)(idx & 63);
  if (e4 >= NE) return;
  const int s = ei[e4], d = ei[NE + e4];
  const v4f ev = *(const v4f*)(E + (size_t)e4 * DIM + c4 * 4);
  const v4f vh = *(const v4f*)(Vh + (size_t)d * DIM + c4 * 4);
  float* ap = agg + (size_t)s * DIM + c4 * 4;
#pragma unroll
  for (int i = 0; i < 4; ++i) {
    const float m = vh[i] * (1.f / (1.f + __expf(-ev[i])));
    atomicAdd(ap + i, m);
  }
}

// ---------------------------------------------------------------------------
// h-branch: x = Uh + agg -> h_out region (scratch), plus BN column stats.
// ---------------------------------------------------------------------------
__global__ void k_hstats(const float* __restrict__ Uh, const float* __restrict__ agg,
                         float* __restrict__ Xh,
                         float* __restrict__ colsum, float* __restrict__ colsq) {
  const int col = threadIdx.x;
  const int rb = blockIdx.x * 100;
  float s = 0.f, q = 0.f;
  for (int r = 0; r < 100; ++r) {
    const int row = rb + r;
    if (row < NN) {
      const size_t o = (size_t)row * DIM + col;
      const float x = Uh[o] + agg[o];
      Xh[o] = x;
      s += x;
      q += x * x;
    }
  }
  atomicAdd(&colsum[col], s);
  atomicAdd(&colsq[col],  q);
}

// ---------------------------------------------------------------------------
// BN finalize (biased variance, matching jnp.var ddof=0).
// ---------------------------------------------------------------------------
__global__ void k_bnfin(const float* __restrict__ csh, const float* __restrict__ cqh,
                        const float* __restrict__ gh,  const float* __restrict__ bh,
                        const float* __restrict__ cse, const float* __restrict__ cqe,
                        const float* __restrict__ gee, const float* __restrict__ bee,
                        float* __restrict__ sch, float* __restrict__ shh,
                        float* __restrict__ sce, float* __restrict__ she) {
  const int t = threadIdx.x;  // 512 threads
  if (t < DIM) {
    const float mean = csh[t] / (float)NN;
    const float var  = cqh[t] / (float)NN - mean * mean;
    const float sc   = gh[t] * rsqrtf(var + 1e-5f);
    sch[t] = sc;
    shh[t] = bh[t] - mean * sc;
  } else {
    const int c = t - DIM;
    const float mean = cse[c] / (float)NE;
    const float var  = cqe[c] / (float)NE - mean * mean;
    const float sc   = gee[c] * rsqrtf(var + 1e-5f);
    sce[c] = sc;
    she[c] = bee[c] - mean * sc;
  }
}

// ---------------------------------------------------------------------------
// In-place finalize: out = base + relu(x*scale + shift); x lives in `out`.
// ---------------------------------------------------------------------------
__global__ void k_out(const float* __restrict__ base, const float* __restrict__ sc,
                      const float* __restrict__ sh, float* __restrict__ out,
                      long long n4) {
  const long long i = (long long)blockIdx.x * blockDim.x + threadIdx.x;
  if (i >= n4) return;
  const int c4 = (int)(i & 63);
  const v4f x = *(const v4f*)(out + i * 4);
  const v4f b = *(const v4f*)(base + i * 4);
  v4f y;
#pragma unroll
  for (int j = 0; j < 4; ++j) {
    const int col = c4 * 4 + j;
    float v = x[j] * sc[col] + sh[col];
    v = (v > 0.f) ? v : 0.f;
    y[j] = b[j] + v;
  }
  *(v4f*)(out + i * 4) = y;
}

// ---------------------------------------------------------------------------
extern "C" void kernel_launch(void* const* d_in, const int* in_sizes, int n_in,
                              void* d_out, int out_size, void* d_ws, size_t ws_size,
                              hipStream_t stream) {
  const float* h   = (const float*)d_in[0];
  const float* e   = (const float*)d_in[1];
  const int*   ei  = (const int*)d_in[2];
  const float* Uw  = (const float*)d_in[3];
  const float* Ub  = (const float*)d_in[4];
  const float* Vw  = (const float*)d_in[5];
  const float* Vb  = (const float*)d_in[6];
  const float* Aw  = (const float*)d_in[7];
  const float* Ab  = (const float*)d_in[8];
  const float* Bw  = (const float*)d_in[9];
  const float* Bb  = (const float*)d_in[10];
  const float* Cw  = (const float*)d_in[11];
  const float* Cb  = (const float*)d_in[12];
  const float* gh  = (const float*)d_in[13];
  const float* bh  = (const float*)d_in[14];
  const float* geo = (const float*)d_in[15];
  const float* beo = (const float*)d_in[16];

  // ---- workspace layout ----
  _Float16* w16 = (_Float16*)d_ws;                              // 5 * 64K halves
  float* f = (float*)((char*)d_ws + (size_t)5 * DIM * DIM * sizeof(_Float16));
  const size_t NSZ = (size_t)NN * DIM;
  float* Uh  = f + 0 * NSZ;
  float* Vh  = f + 1 * NSZ;
  float* Bh  = f + 2 * NSZ;
  float* Ch  = f + 3 * NSZ;
  float* agg = f + 4 * NSZ;
  float* st  = f + 5 * NSZ;   // stats, contiguous after agg
  float *csh = st + 0,    *cqh = st + 256,  *sch = st + 512,  *shh = st + 768;
  float *cse = st + 1024, *cqe = st + 1280, *sce = st + 1536, *she = st + 1792;

  float* outh = (float*)d_out;        // [NN, DIM]
  float* oute = outh + NSZ;           // [NE, DIM]

  // 1) weights fp32 -> f16
  k_cvt_w<<<(5 * DIM * DIM) / 256, 256, 0, stream>>>(Uw, Vw, Aw, Bw, Cw, w16);

  // 2) zero agg + stats (contiguous)
  const int zn = (int)NSZ + 2048;
  k_zero<<<(zn + 255) / 256, 256, 0, stream>>>(agg, zn);

  // 3) node GEMMs (Uh, Vh, Bh, Ch)
  dim3 gn((NN + 127) / 128, 1, 4);
  k_gemm_node<<<gn, 256, 0, stream>>>(h, w16, Ub, Vb, Bb, Cb, Uh, Vh, Bh, Ch);

  // 4) gated messages + scatter-sum
  const long long m4 = (long long)NE * 64;
  k_scatter<<<(unsigned)((m4 + 255) / 256), 256, 0, stream>>>(e, Vh, ei, agg);

  // 5) h pre-BN values + column stats
  k_hstats<<<100, 256, 0, stream>>>(Uh, agg, outh, csh, cqh);

  // 6) edge GEMM fused with gather-add + column stats
  dim3 geg((NE + 127) / 128, 1, 1);
  k_gemm_edge<<<geg, 256, 0, stream>>>(e, w16 + (size_t)2 * DIM * DIM, Ab, ei,
                                       Bh, Ch, oute, cse, cqe);

  // 7) BN scale/shift
  k_bnfin<<<1, 512, 0, stream>>>(csh, cqh, gh, bh, cse, cqe, geo, beo,
                                 sch, shh, sce, she);

  // 8) finalize outputs in place
  const long long h4 = (long long)NN * 64;
  k_out<<<(unsigned)((h4 + 255) / 256), 256, 0, stream>>>(h, sch, shh, outh, h4);
  const long long e4 = (long long)NE * 64;
  k_out<<<(unsigned)((e4 + 255) / 256), 256, 0, stream>>>(e, sce, she, oute, e4);
}
